// GCN_43224550867997
// MI455X (gfx1250) — matrix operand order repaired and checked by reading, not compiled, not run on hardware
//
#include <hip/hip_runtime.h>

typedef __attribute__((ext_vector_type(16))) __bf16 v16bf;
typedef __attribute__((ext_vector_type(8)))  __bf16 v8bf;
typedef __attribute__((ext_vector_type(8)))  float  v8f;

#define NNODES 50000
#define NEDGES 800000
#define FIN    256
#define HID    128
#define NCLS   40
#define NCLSP  48

static __device__ __forceinline__ __bf16 f32_to_bf16(float f) {
    union { float f; unsigned int u; } v; v.f = f;
    unsigned int u = v.u;
    unsigned int r = u + 0x7FFFu + ((u >> 16) & 1u);   // round-to-nearest-even
    unsigned short h = (unsigned short)(r >> 16);
    union { unsigned short s; __bf16 b; } o; o.s = h;
    return o.b;
}

// ---------------- degree / norm ----------------
__global__ void k_deg_init(float* __restrict__ deg) {
    int t = blockIdx.x * blockDim.x + threadIdx.x;
    if (t < NNODES) deg[t] = 1.0f;                       // +1 self loop
}
__global__ void k_deg_count(const int* __restrict__ dst, float* __restrict__ deg) {
    int t = blockIdx.x * blockDim.x + threadIdx.x;
    if (t < NEDGES) atomicAdd(&deg[dst[t]], 1.0f);
}
__global__ void k_dinv(const float* __restrict__ deg, float* __restrict__ dinv) {
    int t = blockIdx.x * blockDim.x + threadIdx.x;
    if (t < NNODES) dinv[t] = rsqrtf(deg[t]);
}

// ---------------- precision conversion ----------------
__global__ void k_f32_to_bf16(const float* __restrict__ in, __bf16* __restrict__ out, int n) {
    int t = blockIdx.x * blockDim.x + threadIdx.x;
    if (t < n) out[t] = f32_to_bf16(in[t]);
}
// W is rows x cols row-major; Wt is colsPad x rows (column-major of W), zero-padded
__global__ void k_transpose_bf16(const float* __restrict__ W, __bf16* __restrict__ Wt,
                                 int rows, int cols, int colsPad) {
    int t = blockIdx.x * blockDim.x + threadIdx.x;
    if (t >= rows * colsPad) return;
    int n = t / rows;          // output row = original column
    int k = t % rows;
    Wt[(size_t)n * rows + k] = (n < cols) ? f32_to_bf16(W[(size_t)k * cols + n]) : f32_to_bf16(0.0f);
}

// ---------------- WMMA GEMM: C[M x tilesN*16] = A[M x K] * Bt[tilesN*16 x K]^T ----------------
// A row-major bf16 (lda = K), Bt column-major-of-B i.e. [n][k] bf16, C f32 (ldc)
__global__ void k_gemm_wmma_bf16(const __bf16* __restrict__ A,
                                 const __bf16* __restrict__ Bt,
                                 float* __restrict__ C,
                                 int K, int ldc, int tilesN) {
    int wave = threadIdx.x >> 5;
    int lane = threadIdx.x & 31;
    int tileM = blockIdx.x;
    int tileN = blockIdx.y * 4 + wave;
    if (tileN >= tilesN) return;                         // whole-wave exit keeps EXEC all-1 for WMMA
    int half = lane >> 4;                                // 0: lanes 0-15, 1: lanes 16-31
    int mn   = lane & 15;
    const __bf16* arow = A  + (size_t)(tileM * 16 + mn) * K;
    const __bf16* brow = Bt + (size_t)(tileN * 16 + mn) * K;
    v8f acc = {};
    for (int k0 = 0; k0 < K; k0 += 32) {
        // A 16x32 bf16 fragment: lanes 0-15 K={0..7,16..23}, lanes 16-31 K={8..15,24..31}
        v8bf alo = *(const v8bf*)(arow + k0 + half * 8);
        v8bf ahi = *(const v8bf*)(arow + k0 + 16 + half * 8);
        // B 32x16 fragment: lane holds column mn, lanes 0-15 K={0..7,16..23}, lanes 16-31 K={8..15,24..31}
        v8bf blo = *(const v8bf*)(brow + k0 + half * 8);
        v8bf bhi = *(const v8bf*)(brow + k0 + 16 + half * 8);
        v16bf a, b;
#pragma unroll
        for (int i = 0; i < 8; ++i) { a[i] = alo[i]; a[8 + i] = ahi[i]; b[i] = blo[i]; b[8 + i] = bhi[i]; }
        acc = __builtin_amdgcn_wmma_f32_16x16x32_bf16(false, a, false, b, (short)0, acc, false, false);
    }
    // C/D layout: lanes 0-15 -> M = 0..7 (VGPR r), N = lane; lanes 16-31 -> M = 8..15, N = lane-16
    int mbase = tileM * 16 + half * 8;
    int col   = tileN * 16 + mn;
#pragma unroll
    for (int r = 0; r < 8; ++r) C[(size_t)(mbase + r) * ldc + col] = acc[r];
}

// ---------------- GCN aggregation ----------------
__global__ void k_init_agg1(const float* __restrict__ xw, const float* __restrict__ dinv,
                            const float* __restrict__ bias, float* __restrict__ agg) {
    int t = blockIdx.x * blockDim.x + threadIdx.x;
    if (t >= NNODES * HID) return;
    int row = t >> 7, col = t & (HID - 1);
    float di = dinv[row];
    agg[t] = xw[t] * di * di + bias[col];
}
__global__ void k_init_agg2(const float* __restrict__ xw, const float* __restrict__ dinv,
                            const float* __restrict__ bias, float* __restrict__ agg) {
    int t = blockIdx.x * blockDim.x + threadIdx.x;
    if (t >= NNODES * NCLS) return;
    int row = t / NCLS, col = t % NCLS;
    float di = dinv[row];
    agg[t] = xw[(size_t)row * NCLSP + col] * di * di + bias[col];
}
// warp per edge: agg[dst] += feat[src] * dinv[src]*dinv[dst]
__global__ void k_edge_scatter(const float* __restrict__ feat, const int* __restrict__ src,
                               const int* __restrict__ dst, const float* __restrict__ dinv,
                               float* __restrict__ agg, int D, int ldf, int lda) {
    int e = (blockIdx.x * blockDim.x + threadIdx.x) >> 5;
    int lane = threadIdx.x & 31;
    if (e >= NEDGES) return;
    int s = src[e], d = dst[e];
    float nrm = dinv[s] * dinv[d];
    const float* f = feat + (size_t)s * ldf;
    float* a = agg + (size_t)d * lda;
    for (int j = lane; j < D; j += 32) atomicAdd(&a[j], f[j] * nrm);
}
__global__ void k_relu_bf16(const float* __restrict__ agg, __bf16* __restrict__ out) {
    int t = blockIdx.x * blockDim.x + threadIdx.x;
    if (t < NNODES * HID) out[t] = f32_to_bf16(fmaxf(agg[t], 0.0f));
}

// ---------------- log_softmax over 40 classes, warp per node ----------------
__global__ void k_logsoftmax(const float* __restrict__ a, float* __restrict__ out) {
    int node = (blockIdx.x * blockDim.x + threadIdx.x) >> 5;
    int lane = threadIdx.x & 31;
    if (node >= NNODES) return;
    const float* row = a + (size_t)node * NCLS;
    float v0 = row[lane];                                     // lane < 32 < 40: always valid
    float v1 = (lane + 32 < NCLS) ? row[lane + 32] : -1e30f;
    float m = fmaxf(v0, v1);
#pragma unroll
    for (int o = 16; o > 0; o >>= 1) m = fmaxf(m, __shfl_xor(m, o, 32));
    float s = expf(v0 - m);
    if (lane + 32 < NCLS) s += expf(v1 - m);
#pragma unroll
    for (int o = 16; o > 0; o >>= 1) s += __shfl_xor(s, o, 32);
    float lse = m + logf(s);
    out[(size_t)node * NCLS + lane] = v0 - lse;
    if (lane + 32 < NCLS) out[(size_t)node * NCLS + lane + 32] = v1 - lse;
}

extern "C" void kernel_launch(void* const* d_in, const int* in_sizes, int n_in,
                              void* d_out, int out_size, void* d_ws, size_t ws_size,
                              hipStream_t stream) {
    const float* x  = (const float*)d_in[0];
    const int*   ei = (const int*)d_in[1];
    const float* W1 = (const float*)d_in[2];
    const float* b1 = (const float*)d_in[3];
    const float* W2 = (const float*)d_in[4];
    const float* b2 = (const float*)d_in[5];
    const int* src = ei;
    const int* dst = ei + NEDGES;

    char* ws = (char*)d_ws;
    size_t off = 0;
    auto alloc = [&](size_t bytes) -> void* {
        void* p = ws + off;
        off = (off + bytes + 255) & ~(size_t)255;
        return p;
    };
    float*  deg   = (float*)alloc((size_t)NNODES * 4);
    float*  dinv  = (float*)alloc((size_t)NNODES * 4);
    __bf16* xbf   = (__bf16*)alloc((size_t)NNODES * FIN * 2);
    __bf16* w1t   = (__bf16*)alloc((size_t)HID * FIN * 2);
    float*  xw1   = (float*)alloc((size_t)NNODES * HID * 4);
    float*  agg1  = (float*)alloc((size_t)NNODES * HID * 4);
    __bf16* h1bf  = (__bf16*)alloc((size_t)NNODES * HID * 2);
    __bf16* w2t   = (__bf16*)alloc((size_t)NCLSP * HID * 2);
    float*  xw2   = (float*)alloc((size_t)NNODES * NCLSP * 4);
    float*  agg2  = (float*)alloc((size_t)NNODES * NCLS * 4);

    const int B = 256;
    // degrees -> dinv
    k_deg_init<<<(NNODES + B - 1) / B, B, 0, stream>>>(deg);
    k_deg_count<<<(NEDGES + B - 1) / B, B, 0, stream>>>(dst, deg);
    k_dinv<<<(NNODES + B - 1) / B, B, 0, stream>>>(deg, dinv);
    // bf16 operands
    k_f32_to_bf16<<<(NNODES * FIN + B - 1) / B, B, 0, stream>>>(x, xbf, NNODES * FIN);
    k_transpose_bf16<<<(FIN * HID + B - 1) / B, B, 0, stream>>>(W1, w1t, FIN, HID, HID);
    k_transpose_bf16<<<(HID * NCLSP + B - 1) / B, B, 0, stream>>>(W2, w2t, HID, NCLS, NCLSP);
    // layer 1: xw1 = x @ W1  (tiles: 3125 x 8)
    k_gemm_wmma_bf16<<<dim3(NNODES / 16, 2), 128, 0, stream>>>(xbf, w1t, xw1, FIN, HID, HID / 16);
    k_init_agg1<<<(NNODES * HID + B - 1) / B, B, 0, stream>>>(xw1, dinv, b1, agg1);
    k_edge_scatter<<<(NEDGES * 32 + B - 1) / B, B, 0, stream>>>(xw1, src, dst, dinv, agg1, HID, HID, HID);
    k_relu_bf16<<<(NNODES * HID + B - 1) / B, B, 0, stream>>>(agg1, h1bf);
    // layer 2: xw2 = relu(h1) @ W2  (tiles: 3125 x 3, padded to 48 cols)
    k_gemm_wmma_bf16<<<dim3(NNODES / 16, 1), 128, 0, stream>>>(h1bf, w2t, xw2, HID, NCLSP, NCLSP / 16);
    k_init_agg2<<<(NNODES * NCLS + B - 1) / B, B, 0, stream>>>(xw2, dinv, b2, agg2);
    k_edge_scatter<<<(NEDGES * 32 + B - 1) / B, B, 0, stream>>>(xw2, src, dst, dinv, agg2, NCLS, NCLSP, NCLS);
    // log_softmax -> d_out
    k_logsoftmax<<<((NNODES + 3) / 4), 128, 0, stream>>>(agg2, (float*)d_out);
}
